// MultiHeadAttention_84104049590613
// MI455X (gfx1250) — matrix operand order repaired and checked by reading, not compiled, not run on hardware
//
#include <hip/hip_runtime.h>
#include <hip/hip_bf16.h>

// ---------------------------------------------------------------------------
// Multi-head attention for MI455X (gfx1250), wave32 + WMMA f16->f32.
//   B=2, S=2048, E=512, H=8, DH=64
// Pipeline:
//   1) cvtT_kernel        : W[E][E] f32 -> Wt[n][k] f16 (transposed, x4)
//   2) gemm_kernel<0/1>   : Q/K = X*W + b  -> f16 [B,H,S,DH]  (Q pre-scaled /8)
//      gemm_kernel<2>     : V   = X*W + b  -> f16 [B,H,DH,S]  (transposed)
//   3) attn_kernel        : flash-attention (online softmax, mask + exp(dm))
//                           -> f16 ctx [B,S,E]
//   4) gemm_kernel<3>     : out = ctx*Wo + bo -> f32 d_out [B,S,E]
// ---------------------------------------------------------------------------

typedef __attribute__((ext_vector_type(16))) _Float16 v16h;
typedef __attribute__((ext_vector_type(8)))  _Float16 v8h;
typedef __attribute__((ext_vector_type(8)))  float    v8f;

#define B_  2
#define S_  2048
#define E_  512
#define H_  8
#define DH_ 64
#define M_  (B_ * S_)          // 4096 GEMM rows

union Frag { v16h v; v8h h[2]; };

static __device__ __forceinline__ v8f wmma_f16(v16h a, v16h b, v8f c) {
  // D = A(16x32 f16) * B(32x16 f16) + C(16x16 f32)
  return __builtin_amdgcn_wmma_f32_16x16x32_f16(
      /*neg_a=*/false, a, /*neg_b=*/false, b,
      /*c_mod=*/(short)0, c, /*reuse_a=*/false, /*reuse_b=*/false);
}

// --------------------------- weight transpose/convert ----------------------
__global__ __launch_bounds__(256) void cvtT_kernel(const float* __restrict__ W,
                                                   _Float16* __restrict__ Wt) {
  int id = blockIdx.x * 256 + threadIdx.x;     // E*E threads
  int k = id >> 9;                             // row of W  (input dim)
  int n = id & (E_ - 1);                       // col of W  (output dim)
  Wt[(size_t)n * E_ + k] = (_Float16)W[id];
}

// --------------------------- WMMA GEMM (4096 x 512 x 512) ------------------
// Wave computes 16(M) x 64(N); block = 8 waves -> 128(M) x 64(N) tile.
// grid = (N/64, M/128) = (8, 32).
// MODE 0: Q head-split f16, pre-scaled by 1/sqrt(DH)
// MODE 1: K head-split f16
// MODE 2: V transposed f16
// MODE 3: f32 output (+bias) to d_out, f16 A input
template <int MODE>
__global__ __launch_bounds__(256) void gemm_kernel(
    const float* __restrict__ Af32, const _Float16* __restrict__ Af16,
    const _Float16* __restrict__ Wt, const float* __restrict__ bias,
    _Float16* __restrict__ out16, float* __restrict__ out32) {
  const int lane = threadIdx.x & 31;
  const int wave = threadIdx.x >> 5;
  const int hi   = lane >> 4;         // 0: lanes 0-15, 1: lanes 16-31
  const int lo   = lane & 15;

  const int mBase = blockIdx.y * 128 + wave * 16;
  const int nBase = blockIdx.x * 64;
  const int row   = mBase + lo;       // A-fragment row (lane = M)
  const int koffA = hi * 8;           // A K-split per half-wave
  const int koffB = hi * 8;           // B K-split per half-wave

  v8f acc[4] = {};

  for (int kk = 0; kk < E_; kk += 32) {
    Frag aF;
    if (MODE == 3) {
      const _Float16* ap = Af16 + (size_t)row * E_ + kk + koffA;
      aF.h[0] = *(const v8h*)(ap);
      aF.h[1] = *(const v8h*)(ap + 16);
    } else {
      const float* ap = Af32 + (size_t)row * E_ + kk + koffA;
      typedef __attribute__((ext_vector_type(8))) float v8fv;
      v8fv f0 = *(const v8fv*)(ap);
      v8fv f1 = *(const v8fv*)(ap + 16);
#pragma unroll
      for (int i = 0; i < 8; ++i) {
        aF.h[0][i] = (_Float16)f0[i];
        aF.h[1][i] = (_Float16)f1[i];
      }
    }
#pragma unroll
    for (int nt = 0; nt < 4; ++nt) {
      const _Float16* bp =
          Wt + (size_t)(nBase + nt * 16 + lo) * E_ + kk + koffB;
      Frag bF;
      bF.h[0] = *(const v8h*)(bp);
      bF.h[1] = *(const v8h*)(bp + 16);
      acc[nt] = wmma_f16(aF.v, bF.v, acc[nt]);
    }
  }

  // Epilogue: bias + layout-specific store.
#pragma unroll
  for (int nt = 0; nt < 4; ++nt) {
    const int ncol = nBase + nt * 16 + lo;     // global N (output feature)
    const float bv = bias[ncol];
#pragma unroll
    for (int r = 0; r < 8; ++r) {
      float val      = acc[nt][r] + bv;
      const int mrow = mBase + r + hi * 8;     // global M
      if (MODE == 3) {
        out32[(size_t)mrow * E_ + ncol] = val;
      } else {
        if (MODE == 0) val *= 0.125f;          // fold 1/sqrt(DH) into Q
        const int b  = mrow >> 11;             // / S_
        const int s  = mrow & (S_ - 1);
        const int h  = ncol >> 6;              // / DH_
        const int dh = ncol & (DH_ - 1);
        size_t idx;
        if (MODE == 2)  // V transposed: [B,H,DH,S]
          idx = (((size_t)b * H_ + h) * DH_ + dh) * S_ + s;
        else            // Q/K head-split: [B,H,S,DH]
          idx = (((size_t)b * H_ + h) * S_ + s) * DH_ + dh;
        out16[idx] = (_Float16)val;
      }
    }
  }
}

// --------------------------- flash attention -------------------------------
// One wave per 16-query tile per (b,h): 2048 waves = 256 blocks x 8 waves.
__global__ __launch_bounds__(256) void attn_kernel(
    const _Float16* __restrict__ Qh, const _Float16* __restrict__ Kh,
    const _Float16* __restrict__ Vt, const int* __restrict__ mask,
    const float* __restrict__ dict_mask, const float* __restrict__ head_w,
    _Float16* __restrict__ ctx) {
  __shared__ _Float16 Plds[8 * 16 * 32];   // per-wave 16x32 f16 P tile (8 KB)

  const int lane  = threadIdx.x & 31;
  const int wave  = threadIdx.x >> 5;
  const int hi    = lane >> 4;
  const int lo    = lane & 15;
  const int gw    = blockIdx.x * 8 + wave;     // 0..2047
  const int qt    = gw & ((S_ / 16) - 1);      // query tile
  const int bh    = gw / (S_ / 16);            // b*H + h
  const int h     = bh & (H_ - 1);
  const int b     = bh >> 3;

  const float hw0 = head_w[h * 2 + 0];
  const float hw1 = head_w[h * 2 + 1];
  const size_t DSZ = (size_t)B_ * S_ * S_;     // dict_mask channel stride

  // Per-lane base pointers (K rows indexed by lo within the key tile).
  const _Float16* Kbase = Kh + ((size_t)bh * S_ + lo) * DH_ + hi * 8;
  const _Float16* Vbase = Vt + ((size_t)bh * DH_ + lo) * S_ + hi * 8;

  // Q fragments for the two DH k-steps (loop invariant). Lane = query row.
  Frag aq[2];
#pragma unroll
  for (int dk = 0; dk < 2; ++dk) {
    const _Float16* qp =
        Qh + ((size_t)bh * S_ + qt * 16 + lo) * DH_ + dk * 32 + hi * 8;
    aq[dk].h[0] = *(const v8h*)(qp);
    aq[dk].h[1] = *(const v8h*)(qp + 16);
  }

  float mrun[8], lrun[8];
#pragma unroll
  for (int r = 0; r < 8; ++r) { mrun[r] = -1.0e30f; lrun[r] = 0.0f; }
  v8f acc[4] = {};                              // O tile 16 x 64 (f32)

  _Float16* myP = &Plds[wave * (16 * 32)];

  for (int kt = 0; kt < S_ / 32; ++kt) {
    const int kbase = kt * 32;

    // Prefetch next key tile's K/V fragments (global_prefetch_b8).
    if (kt + 1 < S_ / 32) {
      const int kn = kbase + 32;
#pragma unroll
      for (int sub = 0; sub < 2; ++sub)
        __builtin_prefetch(Kbase + (size_t)(kn + sub * 16) * DH_, 0, 3);
#pragma unroll
      for (int nt = 0; nt < 4; ++nt)
        __builtin_prefetch(Vbase + (size_t)nt * 16 * S_ + kn, 0, 3);
    }

    // ---- scores S = (Q/8) * K^T for two 16-key subtiles ----------------
    v8f sc[2];
#pragma unroll
    for (int sub = 0; sub < 2; ++sub) {
      v8f z = {};
#pragma unroll
      for (int dk = 0; dk < 2; ++dk) {
        const _Float16* kp =
            Kbase + (size_t)(kbase + sub * 16) * DH_ + dk * 32;
        Frag bk;
        bk.h[0] = *(const v8h*)(kp);
        bk.h[1] = *(const v8h*)(kp + 16);
        z = wmma_f16(aq[dk].v, bk.v, z);
      }
      sc[sub] = z;
    }

    // ---- mask + dict-mask (per element in C layout) --------------------
#pragma unroll
    for (int sub = 0; sub < 2; ++sub) {
      const int kcol = kbase + sub * 16 + lo;
#pragma unroll
      for (int r = 0; r < 8; ++r) {
        const int qrow = qt * 16 + r + hi * 8;
        const size_t mi = ((size_t)b * S_ + qrow) * S_ + kcol;
        float v = sc[sub][r];                   // already scaled by 1/8
        if (mask[mi] == 0) v = -3.0e38f;
        const float dm = hw0 * dict_mask[mi] + hw1 * dict_mask[DSZ + mi];
        v -= __expf(dm);
        sc[sub][r] = v;
      }
    }

    // ---- online softmax: row max/sum via half-wave shfl reductions -----
#pragma unroll
    for (int r = 0; r < 8; ++r) {
      float mx = fmaxf(sc[0][r], sc[1][r]);
      mx = fmaxf(mx, __shfl_xor(mx, 1));
      mx = fmaxf(mx, __shfl_xor(mx, 2));
      mx = fmaxf(mx, __shfl_xor(mx, 4));
      mx = fmaxf(mx, __shfl_xor(mx, 8));
      const float mnew = fmaxf(mrun[r], mx);
      const float resc = __expf(mrun[r] - mnew);
      const float p0 = __expf(sc[0][r] - mnew);
      const float p1 = __expf(sc[1][r] - mnew);
      float rs = p0 + p1;
      rs += __shfl_xor(rs, 1);
      rs += __shfl_xor(rs, 2);
      rs += __shfl_xor(rs, 4);
      rs += __shfl_xor(rs, 8);
      lrun[r] = lrun[r] * resc + rs;
      mrun[r] = mnew;
#pragma unroll
      for (int j = 0; j < 4; ++j) acc[j][r] *= resc;
      // stage P (row-major 16x32) for C-layout -> A-layout transpose
      const int prow = r + hi * 8;
      myP[prow * 32 + lo]      = (_Float16)p0;
      myP[prow * 32 + lo + 16] = (_Float16)p1;
    }

    __syncthreads();   // uniform trip count; publish P before A-layout reads

    // ---- O += P * V  (A = P from LDS, B = V^T contiguous) --------------
    Frag ap;
    const _Float16* pp = myP + lo * 32 + hi * 8;
    ap.h[0] = *(const v8h*)(pp);
    ap.h[1] = *(const v8h*)(pp + 16);
#pragma unroll
    for (int nt = 0; nt < 4; ++nt) {
      const _Float16* vp = Vbase + (size_t)nt * 16 * S_ + kbase;
      Frag bv;
      bv.h[0] = *(const v8h*)(vp);
      bv.h[1] = *(const v8h*)(vp + 16);
      acc[nt] = wmma_f16(ap.v, bv.v, acc[nt]);
    }
  }

  // ---- finalize: divide by l, store f16 ctx [B,S,E] --------------------
#pragma unroll
  for (int nt = 0; nt < 4; ++nt) {
#pragma unroll
    for (int r = 0; r < 8; ++r) {
      const float l   = lrun[r];
      const float inv = (l > 0.0f) ? (1.0f / l) : 0.0f;
      const int qrow  = qt * 16 + r + hi * 8;
      const int dh    = nt * 16 + lo;
      ctx[((size_t)b * S_ + qrow) * E_ + h * DH_ + dh] =
          (_Float16)(acc[nt][r] * inv);
    }
  }
}

// --------------------------- host-side launch ------------------------------
extern "C" void kernel_launch(void* const* d_in, const int* in_sizes, int n_in,
                              void* d_out, int out_size, void* d_ws,
                              size_t ws_size, hipStream_t stream) {
  (void)in_sizes; (void)n_in; (void)out_size; (void)ws_size;
  const float* query = (const float*)d_in[0];
  const float* key   = (const float*)d_in[1];
  const float* value = (const float*)d_in[2];
  const int*   mask  = (const int*)  d_in[3];
  const float* dictm = (const float*)d_in[4];
  const float* wq = (const float*)d_in[5];  const float* bq = (const float*)d_in[6];
  const float* wk = (const float*)d_in[7];  const float* bk = (const float*)d_in[8];
  const float* wv = (const float*)d_in[9];  const float* bv = (const float*)d_in[10];
  const float* wo = (const float*)d_in[11]; const float* bo = (const float*)d_in[12];
  const float* hw = (const float*)d_in[13];
  float* out = (float*)d_out;

  // Workspace layout (f16): 4 x Wt (512 KB) | Qh | Kh | Vt | ctx (4 MB each)
  char* ws = (char*)d_ws;
  const size_t WT  = (size_t)E_ * E_ * sizeof(_Float16);            // 512 KB
  const size_t TEN = (size_t)B_ * H_ * S_ * DH_ * sizeof(_Float16); // 4 MB
  _Float16* Wtq = (_Float16*)(ws);
  _Float16* Wtk = (_Float16*)(ws + WT);
  _Float16* Wtv = (_Float16*)(ws + 2 * WT);
  _Float16* Wto = (_Float16*)(ws + 3 * WT);
  _Float16* Qh  = (_Float16*)(ws + 4 * WT);
  _Float16* Kh  = (_Float16*)(ws + 4 * WT + TEN);
  _Float16* Vt  = (_Float16*)(ws + 4 * WT + 2 * TEN);
  _Float16* ctx = (_Float16*)(ws + 4 * WT + 3 * TEN);

  const dim3 cvtGrid(E_ * E_ / 256);
  cvtT_kernel<<<cvtGrid, 256, 0, stream>>>(wq, Wtq);
  cvtT_kernel<<<cvtGrid, 256, 0, stream>>>(wk, Wtk);
  cvtT_kernel<<<cvtGrid, 256, 0, stream>>>(wv, Wtv);
  cvtT_kernel<<<cvtGrid, 256, 0, stream>>>(wo, Wto);

  const dim3 gGrid(E_ / 64, M_ / 128);   // (8, 32)
  gemm_kernel<0><<<gGrid, 256, 0, stream>>>(query, nullptr, Wtq, bq, Qh, nullptr);
  gemm_kernel<1><<<gGrid, 256, 0, stream>>>(key,   nullptr, Wtk, bk, Kh, nullptr);
  gemm_kernel<2><<<gGrid, 256, 0, stream>>>(value, nullptr, Wtv, bv, Vt, nullptr);

  attn_kernel<<<256, 256, 0, stream>>>(Qh, Kh, Vt, mask, dictm, hw, ctx);

  gemm_kernel<3><<<gGrid, 256, 0, stream>>>(nullptr, ctx, Wto, bo, nullptr, out);
}
